// WideResGEChebNet_50663434224093
// MI455X (gfx1250) — compile-verified, matrix operand order checked by reading, not run on hardware
//
#include <hip/hip_runtime.h>
#include <cstdint>
#include <cmath>

// ---------------------------------------------------------------------------
// WideResGEChebNet for MI455X (gfx1250): bf16 WMMA GEMMs + async-LDS staging
// + L2-resident SpMM scatter
// ---------------------------------------------------------------------------

typedef __attribute__((ext_vector_type(16))) __bf16 bf16x16;
typedef __attribute__((ext_vector_type(8)))  __bf16 bf16x8;
typedef __attribute__((ext_vector_type(8)))  float  f32x8;

#define NB_B    8
#define NB_CIN  3
#define NB_V    4096
#define NB_E    32768
#define NB_K    4
#define NB_NCLS 10
#define NB_M    (NB_V * NB_B)   // 32768 rows (v,b)
#define CMAX    256
#define KDMAX   1024
#define EPSV    1e-5f

__device__ __forceinline__ __bf16 f2bf(float f) {
  union { float f; uint32_t u; } a; a.f = f;
  uint32_t r = a.u + 0x7FFFu + ((a.u >> 16) & 1u);   // round-to-nearest-even
  union { unsigned short s; __bf16 b; } o; o.s = (unsigned short)(r >> 16);
  return o.b;
}

// ----------------------------- tiny utility kernels ------------------------

__global__ void k_zero(uint32_t* __restrict__ p, long long n) {
  long long i = (long long)blockIdx.x * blockDim.x + threadIdx.x;
  if (i < n) p[i] = 0u;
}

// x:(B,CIN,V) -> out:(M,CIN) with row r = v*B+b
__global__ void k_pack_input(const float* __restrict__ x, float* __restrict__ out) {
  long long i = (long long)blockIdx.x * blockDim.x + threadIdx.x;
  if (i >= (long long)NB_M * NB_CIN) return;
  int c = (int)(i % NB_CIN);
  long long r = i / NB_CIN;
  int b = (int)(r % NB_B);
  int v = (int)(r / NB_B);
  out[i] = x[((long long)b * NB_CIN + c) * NB_V + v];
}

// src:(M,Ci) f32 -> A[r*KdPad + kslice*Ci + c] bf16
__global__ void k_pack_a(const float* __restrict__ src, __bf16* __restrict__ A,
                         int Ci, int KdPad, int kslice) {
  long long i = (long long)blockIdx.x * blockDim.x + threadIdx.x;
  if (i >= (long long)NB_M * Ci) return;
  int c = (int)(i % Ci);
  long long r = i / Ci;
  A[r * KdPad + (long long)kslice * Ci + c] = f2bf(src[i]);
}

// w:(Kd,N) f32 -> Wb:(KdPad,N) bf16 (zero-padded rows)
__global__ void k_pack_w(const float* __restrict__ w, __bf16* __restrict__ Wb,
                         int Kd, int KdPad, int N) {
  long long i = (long long)blockIdx.x * blockDim.x + threadIdx.x;
  if (i >= (long long)KdPad * N) return;
  int row = (int)(i / N);
  Wb[i] = (row < Kd) ? f2bf(w[i]) : f2bf(0.0f);
}

// Y[row[e]*F + j] += scale*val[e]*X[col[e]*F + j]  (F = B*Ci, L2-resident atomics)
__global__ void k_spmm(const int* __restrict__ rows, const int* __restrict__ cols,
                       const float* __restrict__ vals,
                       const float* __restrict__ X, float* __restrict__ Y,
                       int F, float scale) {
  int e = blockIdx.x;
  int r = rows[e], c = cols[e];
  float v = vals[e] * scale;
  const float* src = X + (long long)c * F;
  float* dst = Y + (long long)r * F;
  for (int j = threadIdx.x; j < F; j += blockDim.x)
    atomicAdd(&dst[j], v * src[j]);
}

// Chebyshev recurrence combine: y = 2*y - x
__global__ void k_combine(float* __restrict__ y, const float* __restrict__ x, long long n) {
  long long i = (long long)blockIdx.x * blockDim.x + threadIdx.x;
  if (i < n) y[i] = 2.0f * y[i] - x[i];
}

__global__ void k_add(float* __restrict__ dst, const float* __restrict__ a,
                      const float* __restrict__ b, long long n) {
  long long i = (long long)blockIdx.x * blockDim.x + threadIdx.x;
  if (i < n) dst[i] = a[i] + b[i];
}

// Per-channel training-mode BN stats: one block per channel
__global__ void k_bn_stats(const float* __restrict__ X, int C,
                           const float* __restrict__ g, const float* __restrict__ b,
                           float* __restrict__ scale, float* __restrict__ shift) {
  __shared__ float s1[256], s2[256];
  int c = blockIdx.x;
  float a0 = 0.f, a1 = 0.f;
  for (int r = threadIdx.x; r < NB_M; r += 256) {
    float v = X[(long long)r * C + c];
    a0 += v; a1 += v * v;
  }
  s1[threadIdx.x] = a0; s2[threadIdx.x] = a1;
  __syncthreads();
  for (int s = 128; s > 0; s >>= 1) {
    if (threadIdx.x < s) {
      s1[threadIdx.x] += s1[threadIdx.x + s];
      s2[threadIdx.x] += s2[threadIdx.x + s];
    }
    __syncthreads();
  }
  if (threadIdx.x == 0) {
    float m   = s1[0] / (float)NB_M;
    float var = s2[0] / (float)NB_M - m * m;
    float sc  = g[c] * rsqrtf(var + EPSV);
    scale[c] = sc;
    shift[c] = b[c] - m * sc;
  }
}

__global__ void k_bn_relu(const float* __restrict__ X, float* __restrict__ Y,
                          const float* __restrict__ scale, const float* __restrict__ shift,
                          int C, long long n) {
  long long i = (long long)blockIdx.x * blockDim.x + threadIdx.x;
  if (i >= n) return;
  int c = (int)(i % C);
  float v = fmaf(X[i], scale[c], shift[c]);
  Y[i] = v > 0.f ? v : 0.f;
}

// ------------------------------ WMMA GEMM ----------------------------------
// Out(M x N) f32 = A(M x Kd) bf16 * Wb(Kd x N) bf16 (+ bias), Kd % 32 == 0.
// Block tile 128x64, 8 wave32 waves as 4x2, each wave: 2x2 16x16 f32 accums.
// A tile staged via CDNA5 async direct-to-LDS (ASYNCcnt), bypassing VGPRs.
__global__ __launch_bounds__(256) void k_gemm(
    const __bf16* __restrict__ A, const __bf16* __restrict__ Wb,
    const float* __restrict__ bias, float* __restrict__ Out,
    int Kd, int N) {
  __shared__ __align__(16) __bf16 As[128][40];   // 128x32 tile, padded stride
  __shared__ __align__(16) __bf16 Bs[64][40];    // transposed: Bs[col][k]

  int t  = threadIdx.x;
  int m0 = blockIdx.x * 128;
  int n0 = blockIdx.y * 64;
  int wave = t >> 5, lane = t & 31;
  int wm = (wave & 3) * 32;          // wave row offset in tile
  int wn = (wave >> 2) * 32;         // wave col offset in tile
  int lm = lane & 15, h = lane >> 4; // CDNA5 WMMA lane split

  // per-thread A staging assignment: 2 threads/row, 32 contiguous bytes each
  int ar = t >> 1;
  int ac = (t & 1) * 16;
  // generic->LDS: low 32 bits of a shared-aperture address are the LDS offset
  unsigned lds_a = (unsigned)(uintptr_t)&As[ar][ac];

  f32x8 acc[2][2];
  for (int i = 0; i < 2; ++i)
    for (int j = 0; j < 2; ++j)
      for (int v = 0; v < 8; ++v) acc[i][j][v] = 0.0f;

  for (int kt = 0; kt < Kd; kt += 32) {
    // --- stage A tile (128x32 bf16) via async global->LDS copy -----------
    {
      const __bf16* ga = A + (long long)(m0 + ar) * Kd + kt + ac;
      // INST_OFFSET is added to BOTH the LDS and global addresses (ISA 10.x),
      // so one base covers the thread's 32-byte strip with offsets 0 and 16.
      asm volatile(
          "global_load_async_to_lds_b128 %0, %1, off\n\t"
          "global_load_async_to_lds_b128 %0, %1, off offset:16"
          :
          : "v"(lds_a), "v"(ga)
          : "memory");
      if (kt + 32 < Kd)
        __builtin_prefetch(A + (long long)(m0 + ar) * Kd + kt + 32 + ac, 0, 0);
    }
    // --- stage B tile transposed (32x64 -> Bs[col][k]) ---
    {
      int kk = t >> 3;          // 0..31
      int c0 = (t & 7) * 8;     // 0..56
      __bf16 tmp[8];
      if (n0 + c0 < N) {        // N is a multiple of 16, c0 of 8 -> whole-chunk guard
        bf16x8 wv = *(const bf16x8*)(Wb + (long long)(kt + kk) * N + n0 + c0);
        for (int j = 0; j < 8; ++j) tmp[j] = wv[j];
      } else {
        for (int j = 0; j < 8; ++j) tmp[j] = f2bf(0.0f);
      }
      for (int j = 0; j < 8; ++j) Bs[c0 + j][kk] = tmp[j];
    }
    // drain this wave's async copies, then workgroup barrier => tiles visible
    asm volatile("s_wait_asynccnt 0x0" ::: "memory");
    __syncthreads();

    // --- build fragments per ISA 16-bit A 16x32 layout; B mirrored ---
    bf16x16 afrag[2], bfrag[2];
    for (int i = 0; i < 2; ++i) {
      int row = wm + i * 16 + lm;
      bf16x8 lo = *(const bf16x8*)&As[row][h * 8];        // K = h*8 .. +7
      bf16x8 hi = *(const bf16x8*)&As[row][16 + h * 8];   // K = 16+h*8 .. +7
      for (int j = 0; j < 8; ++j) { afrag[i][j] = lo[j]; afrag[i][8 + j] = hi[j]; }
    }
    for (int j = 0; j < 2; ++j) {
      int col = wn + j * 16 + lm;
      bf16x8 lo = *(const bf16x8*)&Bs[col][h * 8];
      bf16x8 hi = *(const bf16x8*)&Bs[col][16 + h * 8];
      for (int q = 0; q < 8; ++q) { bfrag[j][q] = lo[q]; bfrag[j][8 + q] = hi[q]; }
    }

    for (int i = 0; i < 2; ++i)
      for (int j = 0; j < 2; ++j)
        acc[i][j] = __builtin_amdgcn_wmma_f32_16x16x32_bf16(
            false, afrag[i], false, bfrag[j], (short)0, acc[i][j], false, false);

    __syncthreads();
  }

  // --- store: C/D layout: VGPR v -> M = v + h*8, N = lm ---
  for (int i = 0; i < 2; ++i)
    for (int j = 0; j < 2; ++j) {
      int n = n0 + wn + j * 16 + lm;
      if (n >= N) continue;
      float bb = bias ? bias[n] : 0.0f;
      int rbase = m0 + wm + i * 16 + h * 8;
      for (int v = 0; v < 8; ++v)
        Out[(long long)(rbase + v) * N + n] = acc[i][j][v] + bb;
    }
}

// --------------------- head: max-pool over V, FC, log_softmax --------------
__global__ __launch_bounds__(256) void k_head(const float* __restrict__ X,
                                              const float* __restrict__ fw,
                                              const float* __restrict__ fb,
                                              float* __restrict__ out) {
  __shared__ float pooled[NB_B * CMAX];
  __shared__ float logits[NB_B * NB_NCLS];
  int t = threadIdx.x;
  for (int idx = t; idx < NB_B * CMAX; idx += 256) {
    int b = idx / CMAX, c = idx % CMAX;
    float m = -3.4e38f;
    for (int v = 0; v < NB_V; ++v) {
      float val = X[((long long)v * NB_B + b) * CMAX + c];
      m = val > m ? val : m;
    }
    pooled[idx] = m;
  }
  __syncthreads();
  if (t < NB_B * NB_NCLS) {
    int b = t / NB_NCLS, k = t % NB_NCLS;
    float s = fb[k];
    for (int c = 0; c < CMAX; ++c) s += pooled[b * CMAX + c] * fw[c * NB_NCLS + k];
    logits[t] = s;
  }
  __syncthreads();
  if (t < NB_B) {
    float mx = -3.4e38f;
    for (int k = 0; k < NB_NCLS; ++k) mx = fmaxf(mx, logits[t * NB_NCLS + k]);
    float se = 0.f;
    for (int k = 0; k < NB_NCLS; ++k) se += expf(logits[t * NB_NCLS + k] - mx);
    float lse = mx + logf(se);
    for (int k = 0; k < NB_NCLS; ++k)
      out[t * NB_NCLS + k] = logits[t * NB_NCLS + k] - lse;
  }
}

// ------------------------------ host orchestration -------------------------

static inline int cdiv(long long a, long long b) { return (int)((a + b - 1) / b); }

struct Ctx {
  hipStream_t s;
  const int* er; const int* ec; const float* ev;
  __bf16* Abf; __bf16* Wbf;
  float *P0, *P1, *P2;
  float *bnscale, *bnshift;
};

// out(M,Co) = ChebConv(in(M,Ci)) with order Kk (4 or 1 for the shortcut)
static void cheb_conv(const Ctx& C, const float* in, int Ci, int Co, int Kk,
                      const float* w, const float* bias, float* out) {
  int Kd = Kk * Ci;
  int KdPad = (Kd + 31) & ~31;
  long long nw = (long long)KdPad * Co;
  k_pack_w<<<cdiv(nw, 256), 256, 0, C.s>>>(w, C.Wbf, Kd, KdPad, Co);

  if (KdPad != Kd) {
    long long nz = (long long)NB_M * KdPad / 2;   // uint32 count
    k_zero<<<cdiv(nz, 256), 256, 0, C.s>>>((uint32_t*)C.Abf, nz);
  }
  long long nf = (long long)NB_M * Ci;
  k_pack_a<<<cdiv(nf, 256), 256, 0, C.s>>>(in, C.Abf, Ci, KdPad, 0);

  if (Kk > 1) {
    int F = NB_B * Ci;
    // x1 = L x0
    k_zero<<<cdiv(nf, 256), 256, 0, C.s>>>((uint32_t*)C.P1, nf);
    k_spmm<<<NB_E, 256, 0, C.s>>>(C.er, C.ec, C.ev, in, C.P1, F, 1.0f);
    k_pack_a<<<cdiv(nf, 256), 256, 0, C.s>>>(C.P1, C.Abf, Ci, KdPad, 1);
    const float* xm2 = in;
    float* xm1 = C.P1;
    float* tbuf[2] = {C.P2, C.P0};
    for (int k = 2; k < Kk; ++k) {
      float* tb = tbuf[k - 2];
      k_zero<<<cdiv(nf, 256), 256, 0, C.s>>>((uint32_t*)tb, nf);
      k_spmm<<<NB_E, 256, 0, C.s>>>(C.er, C.ec, C.ev, xm1, tb, F, 1.0f);
      k_combine<<<cdiv(nf, 256), 256, 0, C.s>>>(tb, xm2, nf);   // tb = 2*L*x1 - x0
      k_pack_a<<<cdiv(nf, 256), 256, 0, C.s>>>(tb, C.Abf, Ci, KdPad, k);
      xm2 = xm1; xm1 = tb;
    }
  }
  dim3 g(NB_M / 128, (unsigned)cdiv(Co, 64));
  k_gemm<<<g, 256, 0, C.s>>>(C.Abf, C.Wbf, bias, out, KdPad, Co);
}

static void bn_relu(const Ctx& C, const float* X, float* Y, int Cc,
                    const float* g, const float* b) {
  k_bn_stats<<<Cc, 256, 0, C.s>>>(X, Cc, g, b, C.bnscale, C.bnshift);
  long long n = (long long)NB_M * Cc;
  k_bn_relu<<<cdiv(n, 256), 256, 0, C.s>>>(X, Y, C.bnscale, C.bnshift, Cc, n);
}

extern "C" void kernel_launch(void* const* d_in, const int* in_sizes, int n_in,
                              void* d_out, int out_size, void* d_ws, size_t ws_size,
                              hipStream_t stream) {
  const float* x     = (const float*)d_in[0];
  const int*   erows = (const int*)d_in[1];
  const int*   ecols = (const int*)d_in[2];
  const float* evals = (const float*)d_in[3];

  const int HID[4] = {16, 64, 128, 256};

  // jax pytree flattening: dict keys sorted -> 'blocks' first, then conv0_b,
  // conv0_w, fc_b, fc_w; each block dict also key-sorted, sc_w last (l==0 only)
  const float *Bn1b[3][3], *Bn1g[3][3], *Bn2b[3][3], *Bn2g[3][3];
  const float *C1b[3][3],  *C1w[3][3],  *C2b[3][3],  *C2w[3][3],  *Scw[3];
  int idx = 4;
  for (int s = 0; s < 3; ++s)
    for (int l = 0; l < 3; ++l) {
      Bn1b[s][l] = (const float*)d_in[idx++];
      Bn1g[s][l] = (const float*)d_in[idx++];
      Bn2b[s][l] = (const float*)d_in[idx++];
      Bn2g[s][l] = (const float*)d_in[idx++];
      C1b[s][l]  = (const float*)d_in[idx++];
      C1w[s][l]  = (const float*)d_in[idx++];
      C2b[s][l]  = (const float*)d_in[idx++];
      C2w[s][l]  = (const float*)d_in[idx++];
      if (l == 0) Scw[s] = (const float*)d_in[idx++];
    }
  const float* conv0_b = (const float*)d_in[idx++];
  const float* conv0_w = (const float*)d_in[idx++];
  const float* fc_b    = (const float*)d_in[idx++];
  const float* fc_w    = (const float*)d_in[idx++];

  // bump allocator over d_ws (~270 MB total)
  char* ws = (char*)d_ws;
  size_t off = 0;
  auto alloc = [&](size_t bytes) -> void* {
    void* p = ws + off;
    off += (bytes + 255) & ~(size_t)255;
    return p;
  };
  size_t actB = (size_t)NB_M * CMAX * sizeof(float);
  float* ACT0 = (float*)alloc(actB);
  float* ACT1 = (float*)alloc(actB);
  float* Hb   = (float*)alloc(actB);

  Ctx C;
  C.s = stream; C.er = erows; C.ec = ecols; C.ev = evals;
  C.P0 = (float*)alloc(actB);
  C.P1 = (float*)alloc(actB);
  C.P2 = (float*)alloc(actB);
  C.Abf = (__bf16*)alloc((size_t)NB_M * KDMAX * sizeof(__bf16));
  C.Wbf = (__bf16*)alloc((size_t)KDMAX * CMAX * sizeof(__bf16));
  C.bnscale = (float*)alloc(CMAX * sizeof(float));
  C.bnshift = (float*)alloc(CMAX * sizeof(float));

  // input permute (B,CIN,V) -> (M,CIN), then conv0
  k_pack_input<<<cdiv((long long)NB_M * NB_CIN, 256), 256, 0, stream>>>(x, Hb);
  cheb_conv(C, Hb, NB_CIN, HID[0], NB_K, conv0_w, conv0_b, ACT0);

  float* X = ACT0;
  float* Y = ACT1;
  for (int s = 0; s < 3; ++s) {
    for (int l = 0; l < 3; ++l) {
      int ci = (l == 0) ? HID[s] : HID[s + 1];
      int co = HID[s + 1];
      bn_relu(C, X, Hb, ci, Bn1g[s][l], Bn1b[s][l]);              // h = relu(bn1(x))
      cheb_conv(C, Hb, ci, co, NB_K, C1w[s][l], C1b[s][l], Y);    // t = conv1(h)
      bn_relu(C, Y, Y, co, Bn2g[s][l], Bn2b[s][l]);               // u = relu(bn2(t))
      cheb_conv(C, Y, co, co, NB_K, C2w[s][l], C2b[s][l], Y);     // o = conv2(u) (in-place safe)
      long long n = (long long)NB_M * co;
      if (l == 0) {
        cheb_conv(C, Hb, ci, co, 1, Scw[s], nullptr, C.P0);       // sc = K=1 conv(h)
        k_add<<<cdiv(n, 256), 256, 0, stream>>>(X, C.P0, Y, n);   // x' = sc + o
      } else {
        k_add<<<cdiv(n, 256), 256, 0, stream>>>(X, X, Y, n);      // x' = x + o
      }
      // result stays in X for the next block
    }
  }
  k_head<<<1, 256, 0, stream>>>(X, fc_w, fc_b, (float*)d_out);
}
